// MultiHeadGraphAttention_8839042695312
// MI455X (gfx1250) — matrix operand order repaired and checked by reading, not compiled, not run on hardware
//
#include <hip/hip_runtime.h>
#include <hip/hip_bf16.h>

// MI455X / gfx1250, wave32. Matrix work on v_wmma_f32_16x16x32_f16.

typedef __attribute__((ext_vector_type(16))) _Float16 v16h;
typedef __attribute__((ext_vector_type(8)))  float    v8f;

#define DEV __device__ __forceinline__

DEV int lane_id() { return threadIdx.x & 31; }

DEV v8f wmma_f16(v16h a, v16h b, v8f c) {
  // (neg_a, A, neg_b, B, c_mod, C, reuse_a, reuse_b)
  return __builtin_amdgcn_wmma_f32_16x16x32_f16(false, a, false, b, (short)0, c,
                                                false, false);
}

// A fragment (16x32 f16) from row-major f32 src (leading dim ld), tile (row0,k0).
// ISA layout: lane<16 -> K {0..7,16..23}; lane>=16 -> K {8..15,24..31}; M = lane&15.
DEV v16h load_a(const float* src, int ld, int row0, int k0, int kmax) {
  int l = lane_id(); int m = l & 15; int hi = l >> 4;
  v16h a;
#pragma unroll
  for (int t = 0; t < 16; ++t) {
    int vg = t >> 1, half = t & 1;
    int k = ((vg & 4) ? 16 : 0) + hi * 8 + (vg & 3) * 2 + half;
    float v = ((k0 + k) < kmax) ? src[(row0 + m) * ld + k0 + k] : 0.0f;
    a[t] = (_Float16)v;
  }
  return a;
}

// B fragment (32x16) with B[k][n] = S[(n0+n)*ld + k0+k]  (i.e. B = S^T slice; used for X @ W^T)
DEV v16h load_b_t(const float* S, int ld, int n0, int k0, int kmax) {
  int l = lane_id(); int n = l & 15; int hi = l >> 4;
  v16h b;
#pragma unroll
  for (int t = 0; t < 16; ++t) {
    int vg = t >> 1, half = t & 1;
    int k = ((vg & 4) ? 16 : 0) + hi * 8 + (vg & 3) * 2 + half;
    float v = ((k0 + k) < kmax) ? S[(n0 + n) * ld + k0 + k] : 0.0f;
    b[t] = (_Float16)v;
  }
  return b;
}

// B fragment (32x16) with B[k][n] = S[(k0+k)*ld + n0+n]  (row-major source)
DEV v16h load_b_rm(const float* S, int ld, int k0, int n0) {
  int l = lane_id(); int n = l & 15; int hi = l >> 4;
  v16h b;
#pragma unroll
  for (int t = 0; t < 16; ++t) {
    int vg = t >> 1, half = t & 1;
    int k = ((vg & 4) ? 16 : 0) + hi * 8 + (vg & 3) * 2 + half;
    b[t] = (_Float16)S[(k0 + k) * ld + n0 + n];
  }
  return b;
}

// ---------------- small prep kernels ----------------

__global__ void k_zero(float* p, int n) {
  int i = blockIdx.x * blockDim.x + threadIdx.x;
  int stride = gridDim.x * blockDim.x;
  for (; i < n; i += stride) p[i] = 0.0f;
}

// adj[u,w,3] += (sign0, sign1, weight)
__global__ void k_scatter(const int* __restrict__ ei, const float* __restrict__ sgn,
                          const float* __restrict__ wt, float* __restrict__ adj, int E) {
  int e = blockIdx.x * blockDim.x + threadIdx.x;
  if (e >= E) return;
  int u = ei[e], w = ei[E + e];
  int base = (u * 1024 + w) * 3;
  atomicAdd(adj + base + 0, sgn[2 * e]);
  atomicAdd(adj + base + 1, sgn[2 * e + 1]);
  atomicAdd(adj + base + 2, wt[e]);
}

// conv_out[p] = sum_c cw[c]*adj_flat[c*N*N + p] + cb   (faithful to .view(3,n,n))
__global__ void k_conv(const float* __restrict__ adj, const float* __restrict__ cw,
                       const float* __restrict__ cb, float* __restrict__ co) {
  int p = blockIdx.x * blockDim.x + threadIdx.x;
  if (p >= 1024 * 1024) return;
  co[p] = cw[0] * adj[p] + cw[1] * adj[p + 1048576] + cw[2] * adj[p + 2097152] + cb[0];
}

// T[s,c] = (pos_emb(s) @ Wpos^T + bpos)[c], s in [0,512) -- only 512 distinct seq values
__global__ void k_T(const float* __restrict__ Wpos, const float* __restrict__ bpos,
                    float* __restrict__ T) {
  __shared__ float pe[128];
  int s = blockIdx.x;
  int m = threadIdx.x;                       // 128 threads
  float j = (float)(m & 63);
  float invf = powf(10000.0f, -(2.0f * j) / 128.0f);
  float ang = (float)s * invf;
  pe[m] = (m < 64) ? sinf(ang) : cosf(ang);
  __syncthreads();
  float acc = bpos[m];
  const float* wr = Wpos + m * 128;
  for (int q = 0; q < 128; ++q) acc += wr[q] * pe[q];
  T[s * 128 + m] = acc;
}

// Q,K,V = x @ W{q,k,v}^T  -- WMMA, one 16x16 tile per wave
__global__ void k_qkv(const float* __restrict__ x,
                      const float* __restrict__ Wq, const float* __restrict__ Wk,
                      const float* __restrict__ Wv,
                      float* __restrict__ Q, float* __restrict__ K, float* __restrict__ V) {
  int wid = threadIdx.x >> 5;
  int tile = blockIdx.x * 8 + wid;           // 3 * 64 * 8 = 1536 tiles
  int mat = tile / 512, rem = tile % 512;
  int row0 = (rem >> 3) * 16, col0 = (rem & 7) * 16;
  const float* W = (mat == 0) ? Wq : (mat == 1) ? Wk : Wv;
  float* D = (mat == 0) ? Q : (mat == 1) ? K : V;
  v8f c = {};
#pragma unroll
  for (int kc = 0; kc < 4; ++kc) {
    v16h a = load_a(x, 128, row0, kc * 32, 128);
    v16h b = load_b_t(W, 128, col0, kc * 32, 128);
    c = wmma_f16(a, b, c);
  }
  int l = lane_id(); int n = l & 15; int hi = l >> 4;
#pragma unroll
  for (int v = 0; v < 8; ++v)
    D[(row0 + v + 8 * hi) * 128 + col0 + n] = c[v];
}

// Fused: logits(cont+pos+conv) -> softmax -> attn(out) -> out2 = attn @ V_h
// One block per 16 attention rows (g = h*1024 + i). 256 threads = 8 waves.
// Dynamic LDS: L[16][1024] logits/attn tile, Qs[16][16], acc[16][16].
__global__ void k_attn(const float* __restrict__ Q, const float* __restrict__ K,
                       const float* __restrict__ V, const float* __restrict__ T,
                       const float* __restrict__ co, const float* __restrict__ param,
                       const int* __restrict__ seq,
                       float* __restrict__ attn, float* __restrict__ out2) {
  extern __shared__ float sm[];
  float* L   = sm;               // 16*1024
  float* Qs  = sm + 16 * 1024;   // 256
  float* acc = Qs + 256;         // 256

  const int tid = threadIdx.x;
  const int g0 = blockIdx.x * 16;            // 512 blocks
  const int h  = g0 >> 10;
  const int i0 = g0 & 1023;
  const float invs = 0.08838834764831845f;   // 1/sqrt(128)
  const float pc = fmaxf(param[0], 1.0f);    // clamp_min(1) (no-op at 5.0)

  if (tid < 256) { Qs[tid] = Q[g0 * 16 + tid]; acc[tid] = 0.0f; }
  __syncthreads();

  // Phase 1: pos_attn via 512-entry table T + conv term, prescaled, into L.
  // pos[h,a,b] = sum_d Qc[g,d] * T[seq[128h+a/8, (a%8)*128+b/8], (b%8)*16+d]
  for (int idx = tid; idx < 2048; idx += 256) {
    int m = idx >> 7, p = idx & 127;
    int a = i0 + m;
    int r = h * 128 + (a >> 3);
    int s = seq[r * 1024 + (a & 7) * 128 + p];
    const float* Trow = T + s * 128;
    const float* qr = Qs + m * 16;
    const float* cor = co + a * 1024;
#pragma unroll
    for (int t = 0; t < 8; ++t) {
      float d0 = 0.0f;
#pragma unroll
      for (int d = 0; d < 16; ++d) d0 += qr[d] * Trow[t * 16 + d];
      int b = p * 8 + t;
      L[m * 1024 + b] = d0 * invs + pc * cor[b];
    }
  }
  __syncthreads();

  // Phase 2: cont_attn = Qc_tile @ Kc_h^T via WMMA (K=16 zero-padded to 32).
  int wid = tid >> 5, l = tid & 31, n = l & 15, hi = l >> 4;
  v16h aq;
#pragma unroll
  for (int t = 0; t < 16; ++t) {
    int vg = t >> 1, half = t & 1;
    int k = ((vg & 4) ? 16 : 0) + hi * 8 + (vg & 3) * 2 + half;
    aq[t] = (_Float16)((k < 16) ? Qs[(l & 15) * 16 + k] : 0.0f);
  }
  for (int ct = wid; ct < 64; ct += 8) {
    v16h b = load_b_t(K + (h * 1024 + ct * 16) * 16, 16, 0, 0, 16);
    v8f c = {};
    c = wmma_f16(aq, b, c);
#pragma unroll
    for (int v = 0; v < 8; ++v)
      L[(v + 8 * hi) * 1024 + ct * 16 + n] += c[v] * invs;
  }
  __syncthreads();

  // Phase 3: softmax per row (2 rows per wave); write attn to d_out, keep in L.
  for (int mr = wid * 2; mr < wid * 2 + 2; ++mr) {
    float* row = L + mr * 1024;
    float mx = -3.4e38f;
    for (int cI = l; cI < 1024; cI += 32) mx = fmaxf(mx, row[cI]);
#pragma unroll
    for (int o = 16; o; o >>= 1) mx = fmaxf(mx, __shfl_xor(mx, o));
    float sum = 0.0f;
    for (int cI = l; cI < 1024; cI += 32) sum += __expf(row[cI] - mx);
#pragma unroll
    for (int o = 16; o; o >>= 1) sum += __shfl_xor(sum, o);
    float rinv = 1.0f / sum;
    float* arow = attn + (g0 + mr) * 1024;
    for (int cI = l; cI < 1024; cI += 32) {
      float val = __expf(row[cI] - mx) * rinv;
      row[cI] = val;
      arow[cI] = val;
    }
  }
  __syncthreads();

  // Phase 4: out2_tile[16,16] = attn_tile[16,1024] @ Vc_h[1024,16] via WMMA.
  v8f c2 = {};
  for (int kc = wid; kc < 32; kc += 8) {
    v16h a;
#pragma unroll
    for (int t = 0; t < 16; ++t) {
      int vg = t >> 1, half = t & 1;
      int k = ((vg & 4) ? 16 : 0) + hi * 8 + (vg & 3) * 2 + half;
      a[t] = (_Float16)L[(l & 15) * 1024 + kc * 32 + k];
    }
    v16h b = load_b_rm(V + (h * 1024 + kc * 32) * 16, 16, 0, 0);
    c2 = wmma_f16(a, b, c2);
  }
#pragma unroll
  for (int v = 0; v < 8; ++v)
    atomicAdd(&acc[(v + 8 * hi) * 16 + n], c2[v]);
  __syncthreads();
  if (tid < 256) out2[g0 * 16 + tid] = acc[tid];
}

// final: out = out2 (as [1024,128]) @ Wout^T
__global__ void k_out(const float* __restrict__ o2, const float* __restrict__ Wout,
                      float* __restrict__ out) {
  int wid = threadIdx.x >> 5;
  int tile = blockIdx.x * 8 + wid;           // 512 tiles
  int row0 = (tile >> 3) * 16, col0 = (tile & 7) * 16;
  v8f c = {};
#pragma unroll
  for (int kc = 0; kc < 4; ++kc) {
    v16h a = load_a(o2, 128, row0, kc * 32, 128);
    v16h b = load_b_t(Wout, 128, col0, kc * 32, 128);
    c = wmma_f16(a, b, c);
  }
  int l = lane_id(); int n = l & 15; int hi = l >> 4;
#pragma unroll
  for (int v = 0; v < 8; ++v)
    out[(row0 + v + 8 * hi) * 128 + col0 + n] = c[v];
}

extern "C" void kernel_launch(void* const* d_in, const int* in_sizes, int n_in,
                              void* d_out, int out_size, void* d_ws, size_t ws_size,
                              hipStream_t stream) {
  (void)in_sizes; (void)n_in; (void)out_size; (void)ws_size;
  const float* x   = (const float*)d_in[0];
  const float* sgn = (const float*)d_in[1];
  const float* wt  = (const float*)d_in[2];
  const float* Wq  = (const float*)d_in[3];
  const float* Wk  = (const float*)d_in[4];
  const float* Wv  = (const float*)d_in[5];
  const float* Wo  = (const float*)d_in[6];
  const float* Wp  = (const float*)d_in[7];
  const float* bp  = (const float*)d_in[8];
  const float* cw  = (const float*)d_in[9];
  const float* cb  = (const float*)d_in[10];
  const float* pm  = (const float*)d_in[11];
  const int*   seq = (const int*)d_in[12];
  const int*   ei  = (const int*)d_in[13];

  float* out  = (float*)d_out;               // [1024,128]
  float* attn = out + 1024 * 128;            // [8,1024,1024]

  float* ws  = (float*)d_ws;                 // ~18.3 MB used
  float* Q   = ws;                           // 131072
  float* K   = ws + 131072;                  // 131072
  float* V   = ws + 262144;                  // 131072
  float* T   = ws + 393216;                  // 65536
  float* adj = ws + 458752;                  // 3145728
  float* co  = ws + 3604480;                 // 1048576
  float* o2  = ws + 4653056;                 // 131072

  k_zero<<<2048, 256, 0, stream>>>(adj, 3 * 1024 * 1024);
  k_scatter<<<128, 256, 0, stream>>>(ei, sgn, wt, adj, 32768);
  k_conv<<<4096, 256, 0, stream>>>(adj, cw, cb, co);
  k_T<<<512, 128, 0, stream>>>(Wp, bp, T);
  k_qkv<<<192, 256, 0, stream>>>(x, Wq, Wk, Wv, Q, K, V);
  size_t smem = (size_t)(16 * 1024 + 256 + 256) * sizeof(float);  // 67584 B
  k_attn<<<512, 256, smem, stream>>>(Q, K, V, T, co, pm, seq, attn, o2);
  k_out<<<64, 256, 0, stream>>>(o2, Wo, out);
}
